// ActorSvgd_73100343378128
// MI455X (gfx1250) — compile-verified, matrix operand order checked by reading, not run on hardware
//
#include <hip/hip_runtime.h>
#include <hip/hip_bf16.h>
#include <stdint.h>

#define B_   4096
#define N_   32
#define DA   16
#define DO   64
#define H_   256
#define KP   96              // input K padded 80 -> 96
#define BN_  (B_ * N_)
#define LR_  0.1f
#define LIM_ 1.0f

typedef __bf16 bf16t;
typedef bf16t v16bf __attribute__((ext_vector_type(16)));
typedef float v8f  __attribute__((ext_vector_type(8)));

union FragU { uint4 q[2]; v16bf v; };

__device__ __forceinline__ unsigned short f2bf(float x) {
  unsigned u = __float_as_uint(x);
  u += 0x7FFFu + ((u >> 16) & 1u);          // round-to-nearest-even
  return (unsigned short)(u >> 16);
}
__device__ __forceinline__ float bf2f(unsigned short s) {
  return __uint_as_float(((unsigned)s) << 16);
}

// A fragment (16x32 bf16 tile), per ISA 7.12.2: lanes 0-15 -> M=lane, K=hi*8..+7 / +16
__device__ __forceinline__ v16bf loadFragA(const unsigned short* M, int ld,
                                           int mtile, int kc, int lane) {
  int l = lane & 15, hi = lane >> 4;
  const uint4* p = reinterpret_cast<const uint4*>(M + (mtile * 16 + l) * ld + kc * 32 + hi * 8);
  FragU f; f.q[0] = p[0]; f.q[1] = p[2];    // second chunk = +16 elements = +32B
  return f.v;
}
// B fragment (32x16 bf16 tile) from an [N][K]-layout matrix: lane -> column, K contiguous
__device__ __forceinline__ v16bf loadFragB(const unsigned short* Wnk, int ld,
                                           int ntile, int kc, int lane) {
  int l = lane & 15, hi = lane >> 4;
  const uint4* p = reinterpret_cast<const uint4*>(Wnk + (ntile * 16 + l) * ld + kc * 32 + hi * 8);
  FragU f; f.q[0] = p[0]; f.q[1] = p[2];
  return f.v;
}

// ---------------------------------------------------------------------------
// prep: bf16 weight repack ([N][K] layouts), a -> out_a working copy, logp = 0
// ---------------------------------------------------------------------------
__global__ __launch_bounds__(256) void prep_kernel(
    const float* __restrict__ a, const float* __restrict__ w1,
    const float* __restrict__ w2,
    unsigned short* __restrict__ w1t, unsigned short* __restrict__ w2t,
    unsigned short* __restrict__ w2b, unsigned short* __restrict__ w1at,
    float* __restrict__ out_a, float* __restrict__ out_logp) {
  long idx = (long)blockIdx.x * 256 + threadIdx.x;
  const long nA = (long)BN_ * DA;                       // 2,097,152
  if (idx < nA) { out_a[idx] = a[idx]; return; }
  idx -= nA;
  if (idx < BN_) { out_logp[idx] = 0.f; return; }
  idx -= BN_;
  if (idx < 256 * KP) {                                 // w1t[n][k] = w1[k][n], pad K 80->96
    int n = (int)(idx / KP), k = (int)(idx - (long)n * KP);
    w1t[idx] = (k < 80) ? f2bf(w1[k * 256 + n]) : (unsigned short)0;
    return;
  }
  idx -= 256 * KP;
  if (idx < 65536) {                                    // w2t[n][k] = w2[k][n]  (forward B)
    int n = (int)(idx >> 8), k = (int)(idx & 255);
    w2t[idx] = f2bf(w2[k * 256 + n]);
    return;
  }
  idx -= 65536;
  if (idx < 65536) { w2b[idx] = f2bf(w2[idx]); return; } // backward B: (w2^T) in [N][K] = w2 row-major
  idx -= 65536;
  if (idx < 4096) { w1at[idx] = f2bf(w1[64 * 256 + idx]); return; } // rows 64..79 of w1
}

// ---------------------------------------------------------------------------
// one SVGD step, fully fused per batch element (32 particles / workgroup)
// ---------------------------------------------------------------------------
__global__ __launch_bounds__(256) void svgd_step(
    const float* __restrict__ obs, float* __restrict__ a_work,
    const unsigned short* __restrict__ w1t, const unsigned short* __restrict__ w2t,
    const unsigned short* __restrict__ w2b, const unsigned short* __restrict__ w1at,
    const float* __restrict__ w3, const float* __restrict__ b1,
    const float* __restrict__ b2, const float* __restrict__ b3,
    float* __restrict__ out_logp, float* __restrict__ out_q) {
  __shared__ __align__(16) unsigned short sh_h1[32 * 256];   // relu(h1) bf16 (also relu mask)
  __shared__ __align__(16) unsigned short sh_h2[32 * 256];   // relu(h2) bf16, then dh1 bf16
  __shared__ __align__(16) unsigned short sh_dh[32 * 256];   // dh2 bf16
  __shared__ __align__(16) float sh_pool[2048];              // phase1: bf16 act[32*96]; phase2: dist[1024] | K[1024]
  __shared__ float sh_score[32 * 16];
  __shared__ float sh_a[32 * 16];
  __shared__ float sh_q[256];
  __shared__ int sh_cnt;
  __shared__ int sh_imax;

  const int tid = threadIdx.x;
  const int lane = tid & 31;
  const int w = tid >> 5;
  const int l = lane & 15, hi = lane >> 4;
  const int b = blockIdx.x;
  const long rowbase = (long)b * N_;

  unsigned short* sh_act = reinterpret_cast<unsigned short*>(sh_pool);

  // -------- build A for layer 1: [obs | a | 0pad] as bf16, 32 x 96 --------
#pragma unroll
  for (int p = 0; p < 12; ++p) {
    int e = tid + 256 * p;                      // 3072 elements
    int r = e / KP, k = e - r * KP;
    float v;
    if (k < DO)       v = obs[(rowbase + r) * DO + k];
    else if (k < 80)  v = a_work[(rowbase + r) * DA + (k - DO)];
    else              v = 0.f;
    sh_act[e] = f2bf(v);
  }
  __syncthreads();

  const int mt = (w * 4) >> 4;                  // waves 0-3 -> M-tile 0, waves 4-7 -> M-tile 1

  // ---------------- layer 1: h1 = relu(act @ w1 + b1) ----------------
  {
    v8f acc[4] = {};
#pragma unroll
    for (int kc = 0; kc < 3; ++kc) {
      v16bf af = loadFragA(sh_act, KP, mt, kc, lane);
#pragma unroll
      for (int t = 0; t < 4; ++t) {
        v16bf bfr = loadFragB(w1t, KP, (w * 4 + t) & 15, kc, lane);
        acc[t] = __builtin_amdgcn_wmma_f32_16x16x32_bf16(false, af, false, bfr,
                                                         (short)0, acc[t], false, false);
      }
    }
#pragma unroll
    for (int t = 0; t < 4; ++t) {
      int nt = (w * 4 + t) & 15;
#pragma unroll
      for (int g = 0; g < 8; ++g) {
        int m = mt * 16 + hi * 8 + g;
        int n = nt * 16 + l;
        float v = acc[t][g] + b1[n];
        sh_h1[m * 256 + n] = f2bf(v > 0.f ? v : 0.f);
      }
    }
  }
  __syncthreads();

  // ---------------- layer 2: h2 = relu(h1 @ w2 + b2) ----------------
  {
    v8f acc[4] = {};
#pragma unroll
    for (int kc = 0; kc < 8; ++kc) {
      v16bf af = loadFragA(sh_h1, 256, mt, kc, lane);
#pragma unroll
      for (int t = 0; t < 4; ++t) {
        v16bf bfr = loadFragB(w2t, 256, (w * 4 + t) & 15, kc, lane);
        acc[t] = __builtin_amdgcn_wmma_f32_16x16x32_bf16(false, af, false, bfr,
                                                         (short)0, acc[t], false, false);
      }
    }
#pragma unroll
    for (int t = 0; t < 4; ++t) {
      int nt = (w * 4 + t) & 15;
#pragma unroll
      for (int g = 0; g < 8; ++g) {
        int m = mt * 16 + hi * 8 + g;
        int n = nt * 16 + l;
        float v = acc[t][g] + b2[n];
        sh_h2[m * 256 + n] = f2bf(v > 0.f ? v : 0.f);
      }
    }
  }
  __syncthreads();

  // -------- q = h2 @ w3 + b3 ; dh2 = w3 * relu'(h2) --------
  {
    int r = tid >> 3, seg = tid & 7;
    float s = 0.f;
    for (int k = seg * 32; k < seg * 32 + 32; ++k)
      s += bf2f(sh_h2[r * 256 + k]) * w3[k];
    sh_q[tid] = s;
  }
  for (int p = 0; p < 32; ++p) {
    int e = tid + 256 * p;                       // 8192 elements
    int k = e & 255;
    sh_dh[e] = (sh_h2[e] != 0) ? f2bf(w3[k]) : (unsigned short)0;
  }
  __syncthreads();
  if (tid < 32) {
    float s = b3[0];
#pragma unroll
    for (int sg = 0; sg < 8; ++sg) s += sh_q[tid * 8 + sg];
    out_q[rowbase + tid] = s;
  }
  __syncthreads();                               // h2 reads done; safe to overwrite with dh1

  // -------- dh1 = (dh2 @ w2^T) * relu'(h1)  (B = w2 row-major = [N][K] of w2^T) --------
  {
    v8f acc[4] = {};
#pragma unroll
    for (int kc = 0; kc < 8; ++kc) {
      v16bf af = loadFragA(sh_dh, 256, mt, kc, lane);
#pragma unroll
      for (int t = 0; t < 4; ++t) {
        v16bf bfr = loadFragB(w2b, 256, (w * 4 + t) & 15, kc, lane);
        acc[t] = __builtin_amdgcn_wmma_f32_16x16x32_bf16(false, af, false, bfr,
                                                         (short)0, acc[t], false, false);
      }
    }
#pragma unroll
    for (int t = 0; t < 4; ++t) {
      int nt = (w * 4 + t) & 15;
#pragma unroll
      for (int g = 0; g < 8; ++g) {
        int m = mt * 16 + hi * 8 + g;
        int idx = m * 256 + nt * 16 + l;
        sh_h2[idx] = (sh_h1[idx] != 0) ? f2bf(acc[t][g]) : (unsigned short)0;
      }
    }
  }
  __syncthreads();

  // -------- score = dh1 @ w1_a^T   (M=32, N=16, K=256) --------
  if (w < 2) {
    v8f acc = {};
#pragma unroll
    for (int kc = 0; kc < 8; ++kc) {
      v16bf af = loadFragA(sh_h2, 256, w, kc, lane);
      v16bf bfr = loadFragB(w1at, 256, 0, kc, lane);
      acc = __builtin_amdgcn_wmma_f32_16x16x32_bf16(false, af, false, bfr,
                                                    (short)0, acc, false, false);
    }
#pragma unroll
    for (int g = 0; g < 8; ++g) {
      int m = w * 16 + hi * 8 + g;
      sh_score[m * 16 + l] = acc[g];
    }
  }
  if (tid == 0) { sh_imax = 0; sh_cnt = 0; }
  sh_a[tid] = a_work[rowbase * DA + tid];
  sh_a[tid + 256] = a_work[rowbase * DA + tid + 256];
  __syncthreads();

  // ---------------- SVGD phase (32 particles, D=16) ----------------
  float* dist = sh_pool;
  float* Km = sh_pool + 1024;
  float lmax = 0.f;
#pragma unroll
  for (int p = 0; p < 4; ++p) {
    int e = tid + 256 * p;                       // 1024 (i,j) pairs
    int i = e >> 5, j = e & 31;
    float s = 0.f;
#pragma unroll
    for (int d = 0; d < 16; ++d) {
      float df = sh_a[i * 16 + d] - sh_a[j * 16 + d];
      s += df * df;
    }
    dist[e] = s;
    lmax = fmaxf(lmax, s);
  }
  atomicMax(&sh_imax, __float_as_int(lmax));     // non-negative floats: int-compare OK
  __syncthreads();

  // median of the 1024 distances via bisection on a shared counter
  float hi_v = __int_as_float(sh_imax), lo_v = 0.f;
  for (int it = 0; it < 24; ++it) {
    if (tid == 0) sh_cnt = 0;
    __syncthreads();
    float mid = 0.5f * (lo_v + hi_v);
    int c = 0;
#pragma unroll
    for (int p = 0; p < 4; ++p) c += (dist[tid + 256 * p] <= mid) ? 1 : 0;
    atomicAdd(&sh_cnt, c);
    __syncthreads();
    if (sh_cnt >= 512) hi_v = mid; else lo_v = mid;
    __syncthreads();
  }
  float med = 0.5f * (lo_v + hi_v);
  float hband = med / 3.4657359028f + 1e-8f;     // log(32)
  float gamma = 1.f / (2.f * hband);

#pragma unroll
  for (int p = 0; p < 4; ++p) {
    int e = tid + 256 * p;
    Km[e] = __expf(-gamma * dist[e]);
  }
  __syncthreads();

  // phi = (K @ S - sum_j K_grad) / N ; a += LR*phi ; clamp
#pragma unroll
  for (int p = 0; p < 2; ++p) {
    int e = tid + 256 * p;                       // 512 (i,d) slots
    int i = e >> 4, d = e & 15;
    float ai = sh_a[i * 16 + d];
    float drive = 0.f, kg = 0.f;
    for (int j = 0; j < 32; ++j) {
      float kij = Km[i * 32 + j];
      drive += kij * sh_score[j * 16 + d];
      kg += kij * (ai - sh_a[j * 16 + d]);       // sum_j K_grad = -2*gamma*kg
    }
    float phi = (drive + 2.f * gamma * kg) * (1.f / (float)N_);
    float an = ai + LR_ * phi;
    an = fminf(LIM_, fmaxf(-LIM_, an));
    a_work[rowbase * DA + e] = an;
  }

  // logp correction: logp -= LR*(tmp1 + tmp2)
  if (tid < 32) {
    int i = tid;
    float t1 = 0.f, t2 = 0.f;
    for (int j = 0; j < 32; ++j) {
      float kij = Km[i * 32 + j];
      float dot = 0.f;
#pragma unroll
      for (int d = 0; d < 16; ++d)
        dot += (sh_a[i * 16 + d] - sh_a[j * 16 + d]) * sh_score[j * 16 + d];
      t1 += kij * dot;                            // * (-2*gamma)/N later
      t2 += kij * (2.f * gamma * dist[i * 32 + j] - (float)DA);
    }
    float ng = -2.f * gamma / (float)N_;
    out_logp[(long)b * N_ + i] += -LR_ * (ng * t1 + ng * t2);
  }
}

// ---------------------------------------------------------------------------
extern "C" void kernel_launch(void* const* d_in, const int* in_sizes, int n_in,
                              void* d_out, int out_size, void* d_ws, size_t ws_size,
                              hipStream_t stream) {
  (void)in_sizes; (void)n_in; (void)out_size; (void)ws_size;
  const float* obs = (const float*)d_in[0];
  const float* a   = (const float*)d_in[1];
  const float* w1  = (const float*)d_in[2];
  const float* b1  = (const float*)d_in[3];
  const float* w2  = (const float*)d_in[4];
  const float* b2  = (const float*)d_in[5];
  const float* w3  = (const float*)d_in[6];
  const float* b3  = (const float*)d_in[7];

  float* out = (float*)d_out;
  float* out_a    = out;                          // [B,N,16]
  float* out_logp = out + (long)BN_ * DA;         // [B,N]
  float* out_q    = out_logp + BN_;               // [BN]

  char* ws = (char*)d_ws;
  unsigned short* w1t  = (unsigned short*)ws;                       // 256*96 bf16
  unsigned short* w2t  = (unsigned short*)(ws + 49152);             // 256*256 bf16
  unsigned short* w2b  = (unsigned short*)(ws + 49152 + 131072);    // 256*256 bf16
  unsigned short* w1at = (unsigned short*)(ws + 49152 + 2 * 131072);// 16*256 bf16

  const long prep_total = (long)BN_ * DA + BN_ + 256 * KP + 65536 + 65536 + 4096;
  int prep_blocks = (int)((prep_total + 255) / 256);
  prep_kernel<<<prep_blocks, 256, 0, stream>>>(a, w1, w2, w1t, w2t, w2b, w1at,
                                               out_a, out_logp);
  for (int s = 0; s < 3; ++s)
    svgd_step<<<B_, 256, 0, stream>>>(obs, out_a, w1t, w2t, w2b, w1at,
                                      w3, b1, b2, b3, out_logp, out_q);
}
